// MultiAgentActorCritic_80324478370390
// MI455X (gfx1250) — matrix operand order repaired and checked by reading, not compiled, not run on hardware
//
#include <hip/hip_runtime.h>

#define NNODES 2048
#define BATCH  4
#define HEADS  4
#define ROWS   8192   // BATCH * NNODES
#define NEG_INF (-__builtin_inff())

typedef __attribute__((ext_vector_type(16))) __bf16 v16bf;
typedef __attribute__((ext_vector_type(8)))  float  v8f;

union Frag16 { v16bf v; unsigned short u[16]; unsigned int d[8]; };

__device__ __forceinline__ unsigned short f2bf(float f) {
  unsigned int u = __float_as_uint(f);
  u += 0x7FFFu + ((u >> 16) & 1u);   // round-to-nearest-even
  return (unsigned short)(u >> 16);
}

__device__ __forceinline__ v8f vzero8() {
  v8f z;
#pragma unroll
  for (int i = 0; i < 8; ++i) z[i] = 0.0f;
  return z;
}

// ---------------------------------------------------------------------------
// causal^T = transpose(softmax(CS, axis=-1)) emitted as bf16 COLUMN-MAJOR
// (out[c*128 + r]) so GEMM B-fragments are contiguous b128 loads.
// ---------------------------------------------------------------------------
__global__ void k_cs_softmax(const float* __restrict__ CS, unsigned short* __restrict__ outT) {
  __shared__ float red[128];
  int r = blockIdx.x, c = threadIdx.x;
  float v = CS[r * 128 + c];
  red[c] = v;
  __syncthreads();
  for (int s = 64; s > 0; s >>= 1) { if (c < s) red[c] = fmaxf(red[c], red[c + s]); __syncthreads(); }
  float mx = red[0];
  __syncthreads();
  float e = __expf(v - mx);
  red[c] = e;
  __syncthreads();
  for (int s = 64; s > 0; s >>= 1) { if (c < s) red[c] += red[c + s]; __syncthreads(); }
  float inv = 1.0f / red[0];
  outT[c * 128 + r] = f2bf(e * inv);   // transposed store
}

__global__ void k_cast_bf16(const float* __restrict__ in, unsigned short* __restrict__ out, int n) {
  int i = blockIdx.x * blockDim.x + threadIdx.x;
  if (i < n) out[i] = f2bf(in[i]);
}

// Cast + per-head transpose: in [H][K][N] f32 row-major -> out [H][N][K] bf16.
__global__ void k_cast_transpose_bf16(const float* __restrict__ in, unsigned short* __restrict__ out,
                                      int K, int N, int total) {
  int gid = blockIdx.x * blockDim.x + threadIdx.x;
  if (gid >= total) return;
  int kn = K * N;
  int h = gid / kn;
  int rem = gid - h * kn;
  int k = rem / N;
  int n = rem - k * N;
  out[(size_t)h * kn + (size_t)n * K + k] = f2bf(in[gid]);
}

// ---------------------------------------------------------------------------
// bf16 WMMA GEMM: C[M,N] = A[M,K] * B[K,N], A row-major, Bt COLUMN-major
// (Bt[n*K+k]). One wave computes a 16x64 tile. N in {64,128}, K % 32 == 0.
// Both A and B fragments are contiguous dword loads -> global_load_b128.
// ---------------------------------------------------------------------------
__global__ __launch_bounds__(256) void k_gemm_bf16(
    const unsigned short* __restrict__ A, const unsigned short* __restrict__ Bt,
    int M, int N, int K,
    float* __restrict__ outF, unsigned short* __restrict__ outB) {
  int wave = (blockIdx.x * blockDim.x + threadIdx.x) >> 5;
  int lane = threadIdx.x & 31;
  int half = lane >> 4;
  int lrow = lane & 15;
  int tiles_n = N >> 6;
  int tiles = (M >> 4) * tiles_n;
  if (wave >= tiles) return;
  int tm = wave / tiles_n, tn = wave - tm * tiles_n;
  int i0 = tm << 4;
  int n0b = tn << 6;

  v8f acc[4];
#pragma unroll
  for (int t = 0; t < 4; ++t) acc[t] = vzero8();

  const unsigned int* ar = (const unsigned int*)(A + (size_t)(i0 + lrow) * K);
  const unsigned int* bc[4];
#pragma unroll
  for (int t = 0; t < 4; ++t)
    bc[t] = (const unsigned int*)(Bt + (size_t)(n0b + t * 16 + lrow) * K);

  for (int kb = 0; kb < K; kb += 32) {
    int kd = kb >> 1;
    // A fragment (16x32 bf16): lanes 0-15 hold K 0..15, lanes 16-31 hold K 16..31
    Frag16 af;
#pragma unroll
    for (int v = 0; v < 4; ++v) {
      af.d[v]     = ar[kd + half * 8 + v];
      af.d[4 + v] = ar[kd + half * 8 + 4 + v];
    }
#pragma unroll
    for (int t = 0; t < 4; ++t) {
      // B fragment (32x16): lane col = lrow, K base = half*16; pairs contiguous in Bt
      Frag16 bf;
#pragma unroll
      for (int v = 0; v < 8; ++v) bf.d[v] = bc[t][kd + half * 8 + v];
      acc[t] = __builtin_amdgcn_wmma_f32_16x16x32_bf16(false, af.v, false, bf.v,
                                                       (short)0, acc[t], false, false);
    }
  }
  // C layout: lane L holds rows v + 8*(L/16), col = L%16 (per 16-col subtile)
#pragma unroll
  for (int t = 0; t < 4; ++t)
#pragma unroll
    for (int v = 0; v < 8; ++v) {
      int row = i0 + v + 8 * half;
      int col = n0b + t * 16 + lrow;
      float x = acc[t][v];
      if (outF) outF[(size_t)row * N + col] = x;
      if (outB) outB[(size_t)row * N + col] = f2bf(x);
    }
}

// ---------------------------------------------------------------------------
// s_i / s_j score vectors: s_i[h,m] = xt[h,m,:] . a[h,:64], s_j uses a[h,64:].
// ---------------------------------------------------------------------------
__global__ void k_scores(const float* __restrict__ xt, const float* __restrict__ a,
                         float* __restrict__ si, float* __restrict__ sj) {
  int gid = blockIdx.x * blockDim.x + threadIdx.x;
  if (gid >= HEADS * ROWS) return;
  int head = gid >> 13;
  const float* row = xt + (size_t)gid * 64;
  const float* av  = a + head * 128;
  float s0 = 0.f, s1 = 0.f;
#pragma unroll 8
  for (int d = 0; d < 64; ++d) { s0 += row[d] * av[d]; s1 += row[d] * av[64 + d]; }
  si[gid] = s0;
  sj[gid] = s1;
}

// ---------------------------------------------------------------------------
// Flash-GAT attention for one layer. Grid: B*H*(N/128) blocks of 256 threads
// (8 waves); each wave owns 16 query rows. V (= xt bf16) staged to LDS in
// 256-row chunks, stored transposed [64 feat][256 rows] so B fragments are
// contiguous dword DS reads. Online softmax with shfl_xor(16) row reductions.
// Writes to [B*N, 256] at column head*64 (f32 and/or elu->bf16).
// ---------------------------------------------------------------------------
__global__ __launch_bounds__(256) void k_flash_gat(
    const unsigned short* __restrict__ xtb,
    const float* __restrict__ si, const float* __restrict__ sj,
    const int* __restrict__ adj,
    float* __restrict__ outF, unsigned short* __restrict__ outB, int do_elu) {
  __shared__ __align__(16) unsigned short Vlds[64 * 256];
  __shared__ float sjlds[256];

  int b    = blockIdx.x >> 6;
  int rem  = blockIdx.x & 63;
  int head = rem >> 4;
  int iblk = rem & 15;
  int w    = threadIdx.x >> 5;
  int lane = threadIdx.x & 31;
  int half = lane >> 4;
  int lrow = lane & 15;
  int i0   = iblk * 128 + w * 16;

  size_t hb = (size_t)head * ROWS + (size_t)b * NNODES;
  const unsigned short* vbase = xtb + hb * 64;
  const float* sjb = sj + hb;
  float s_i_l = si[hb + i0 + lrow];                       // dst-score for this lane's row
  const int* arow = adj + ((size_t)b * NNODES + i0 + lrow) * (size_t)NNODES;

  // chunk-loader decomposition: thread handles row pair (2*r2, 2*r2+1), col half ch
  int r2 = threadIdx.x & 127;
  int ch = threadIdx.x >> 7;
  unsigned int* V32 = (unsigned int*)Vlds;

  float m = NEG_INF, l = 0.0f;
  v8f acc[4];
#pragma unroll
  for (int t = 0; t < 4; ++t) acc[t] = vzero8();

  for (int cb = 0; cb < NNODES; cb += 256) {
    __syncthreads();
    {   // cooperative transposed chunk load: 32 dword global loads + 32 dword DS stores
      const unsigned int* src0 = (const unsigned int*)(vbase + (size_t)(cb + 2 * r2) * 64) + ch * 16;
      const unsigned int* src1 = (const unsigned int*)(vbase + (size_t)(cb + 2 * r2 + 1) * 64) + ch * 16;
#pragma unroll
      for (int j = 0; j < 16; ++j) {
        unsigned int w0 = src0[j];                        // cols c0,c0+1 of row 2*r2
        unsigned int w1 = src1[j];                        // cols c0,c0+1 of row 2*r2+1
        int c0 = 2 * (j + 16 * ch);
        V32[(c0 * 256 + 2 * r2) >> 1]       = (w0 & 0xffffu) | (w1 << 16);
        V32[((c0 + 1) * 256 + 2 * r2) >> 1] = (w0 >> 16) | (w1 & 0xffff0000u);
      }
      sjlds[threadIdx.x] = sjb[cb + threadIdx.x];
    }
    __syncthreads();

    for (int jt = 0; jt < 256; jt += 32) {
      int j0 = cb + jt;
      // scores in A-fragment K order: u[0..7] -> K=half*8+i, u[8..15] -> K=16+half*8+i
      float sc[16];
      const int*   a1p = arow + j0 + half * 8;
      const int*   a2p = arow + j0 + 16 + half * 8;
      const float* sj1 = sjlds + jt + half * 8;
      const float* sj2 = sjlds + jt + 16 + half * 8;
#pragma unroll
      for (int i = 0; i < 8; ++i) {
        float s = s_i_l + sj1[i];
        s = (s >= 0.f) ? s : 0.2f * s;                    // leaky_relu(0.2)
        sc[i] = (a1p[i] != 0) ? s : NEG_INF;
        float s2 = s_i_l + sj2[i];
        s2 = (s2 >= 0.f) ? s2 : 0.2f * s2;
        sc[8 + i] = (a2p[i] != 0) ? s2 : NEG_INF;
      }
      float lm = NEG_INF;
#pragma unroll
      for (int i = 0; i < 16; ++i) lm = fmaxf(lm, sc[i]);
      float rm = fmaxf(lm, __shfl_xor(lm, 16, 32));       // full-row max (row split over 2 lanes)
      float mn = fmaxf(m, rm);
      bool  ninf  = (mn == NEG_INF);                      // branchless -> EXEC stays all-ones
      float mneff = ninf ? 0.0f : mn;
      float scale = ninf ? 1.0f : __expf(m - mn);

      Frag16 pf;
      float rsum = 0.0f;
#pragma unroll
      for (int i = 0; i < 16; ++i) {
        float p = __expf(sc[i] - mneff);                  // -inf -> 0 naturally
        rsum += p;
        pf.u[i] = f2bf(p);
      }
      rsum += __shfl_xor(rsum, 16, 32);
      l = l * scale + rsum;
      m = mn;

      // rescale accumulators: acc row index = v + 8*half, its scale lives in lane (v+8*half)
      float rowscale[8];
#pragma unroll
      for (int v = 0; v < 8; ++v) rowscale[v] = __shfl(scale, v + 8 * half, 32);
#pragma unroll
      for (int t = 0; t < 4; ++t)
#pragma unroll
        for (int v = 0; v < 8; ++v) acc[t][v] *= rowscale[v];

      // P(16x32) @ V(32x64): 4 WMMAs, B fragments from transposed LDS (contiguous dwords)
#pragma unroll
      for (int t = 0; t < 4; ++t) {
        Frag16 vf;
        const unsigned int* vp =
            (const unsigned int*)(Vlds + (size_t)(t * 16 + lrow) * 256 + jt + half * 8);
#pragma unroll
        for (int v = 0; v < 8; ++v) vf.d[v] = vp[(v >> 2) * 8 + (v & 3)];
        acc[t] = __builtin_amdgcn_wmma_f32_16x16x32_bf16(false, pf.v, false, vf.v,
                                                         (short)0, acc[t], false, false);
      }
    }
  }

  // normalize by l and write out at column head*64 of a 256-wide row
  float linv = 1.0f / l;
  float rinv[8];
#pragma unroll
  for (int v = 0; v < 8; ++v) rinv[v] = __shfl(linv, v + 8 * half, 32);
#pragma unroll
  for (int t = 0; t < 4; ++t)
#pragma unroll
    for (int v = 0; v < 8; ++v) {
      float x = acc[t][v] * rinv[v];
      size_t off = ((size_t)b * NNODES + i0 + v + 8 * half) * 256 + head * 64 + t * 16 + lrow;
      if (outF) outF[off] = x;
      if (outB) {
        float e = do_elu ? ((x > 0.f) ? x : (__expf(x) - 1.0f)) : x;
        outB[off] = f2bf(e);
      }
    }
}

// ---------------------------------------------------------------------------
extern "C" void kernel_launch(void* const* d_in, const int* in_sizes, int n_in,
                              void* d_out, int out_size, void* d_ws, size_t ws_size,
                              hipStream_t stream) {
  const float* x   = (const float*)d_in[0];
  const int*   adj = (const int*)d_in[1];
  const float* CS  = (const float*)d_in[2];
  const float* W0  = (const float*)d_in[3];
  const float* a0  = (const float*)d_in[4];
  const float* W1  = (const float*)d_in[5];
  const float* a1  = (const float*)d_in[6];

  char* ws = (char*)d_ws;
  size_t o = 0;
  auto alloc = [&](size_t bytes) { void* p = ws + o; o += (bytes + 255) & ~(size_t)255; return p; };

  unsigned short* causalT = (unsigned short*)alloc(128 * 128 * 2);         // col-major
  unsigned short* xb      = (unsigned short*)alloc((size_t)ROWS * 128 * 2);
  unsigned short* w0t     = (unsigned short*)alloc((size_t)HEADS * 128 * 64 * 2);  // [H][64][128]
  unsigned short* w1t     = (unsigned short*)alloc((size_t)HEADS * 256 * 64 * 2);  // [H][64][256]
  unsigned short* h0b     = (unsigned short*)alloc((size_t)ROWS * 128 * 2);
  float*          xt0f    = (float*)alloc((size_t)HEADS * ROWS * 64 * 4);
  unsigned short* xt0b    = (unsigned short*)alloc((size_t)HEADS * ROWS * 64 * 2);
  float*          si0     = (float*)alloc((size_t)HEADS * ROWS * 4);
  float*          sj0     = (float*)alloc((size_t)HEADS * ROWS * 4);
  unsigned short* h1b     = (unsigned short*)alloc((size_t)ROWS * 256 * 2);
  float*          xt1f    = (float*)alloc((size_t)HEADS * ROWS * 64 * 4);
  unsigned short* xt1b    = (unsigned short*)alloc((size_t)HEADS * ROWS * 64 * 2);
  float*          si1     = (float*)alloc((size_t)HEADS * ROWS * 4);
  float*          sj1     = (float*)alloc((size_t)HEADS * ROWS * 4);

  // Stage A: causal mask (transposed bf16) + bf16 casts / weight transposes
  k_cs_softmax<<<128, 128, 0, stream>>>(CS, causalT);
  int nx = ROWS * 128;
  k_cast_bf16<<<(nx + 255) / 256, 256, 0, stream>>>(x, xb, nx);
  int n0 = HEADS * 128 * 64;
  k_cast_transpose_bf16<<<(n0 + 255) / 256, 256, 0, stream>>>(W0, w0t, 128, 64, n0);
  int n1 = HEADS * 256 * 64;
  k_cast_transpose_bf16<<<(n1 + 255) / 256, 256, 0, stream>>>(W1, w1t, 256, 64, n1);

  // Stage B: h0 = x @ causal  [8192,128]x[128,128] -> bf16
  k_gemm_bf16<<<(512 * 2) / 8, 256, 0, stream>>>(xb, causalT, ROWS, 128, 128, nullptr, h0b);

  // Layer 0: per-head projection, scores, flash attention (+ELU, head concat)
  for (int h = 0; h < HEADS; ++h)
    k_gemm_bf16<<<512 / 8, 256, 0, stream>>>(h0b, w0t + (size_t)h * 128 * 64, ROWS, 64, 128,
                                             xt0f + (size_t)h * ROWS * 64,
                                             xt0b + (size_t)h * ROWS * 64);
  k_scores<<<(HEADS * ROWS) / 256, 256, 0, stream>>>(xt0f, a0, si0, sj0);
  k_flash_gat<<<BATCH * HEADS * (NNODES / 128), 256, 0, stream>>>(xt0b, si0, sj0, adj,
                                                                  nullptr, h1b, 1);

  // Layer 1: per-head projection from concat'd h1 (K=256), scores, flash, f32 out
  for (int h = 0; h < HEADS; ++h)
    k_gemm_bf16<<<512 / 8, 256, 0, stream>>>(h1b, w1t + (size_t)h * 256 * 64, ROWS, 64, 256,
                                             xt1f + (size_t)h * ROWS * 64,
                                             xt1b + (size_t)h * ROWS * 64);
  k_scores<<<(HEADS * ROWS) / 256, 256, 0, stream>>>(xt1f, a1, si1, sj1);
  k_flash_gat<<<BATCH * HEADS * (NNODES / 128), 256, 0, stream>>>(xt1b, si1, sj1, adj,
                                                                  (float*)d_out, nullptr, 0);
}
